// MHA_7103875908173
// MI455X (gfx1250) — matrix operand order repaired and checked by reading, not compiled, not run on hardware
//
#include <hip/hip_runtime.h>
#include <hip/hip_bf16.h>

typedef __attribute__((ext_vector_type(16))) _Float16 v16h;
typedef __attribute__((ext_vector_type(8)))  float    v8f;

constexpr int BB = 4;        // batch
constexpr int SS = 2048;     // seq
constexpr int EE = 2048;     // embed
constexpr int HH = 16;       // heads
constexpr int DD = 128;      // head dim
constexpr int N3 = 3 * EE;   // 6144
constexpr int KC = EE / 32;  // 64 k-chunks of 32 (all GEMMs have K = EE)
constexpr int MT = (BB * SS) / 16; // 512 row tiles

#define DEV static __device__ __forceinline__

DEV v8f wmma_f16(v16h a, v16h b, v8f c) {
    return __builtin_amdgcn_wmma_f32_16x16x32_f16(false, a, false, b, (short)0, c, false, false);
}

// CDNA5 async global->LDS copy of 16 bytes (GLOBAL_LOAD_ASYNC_TO_LDS_B128).
DEV void async_copy_b128(const void* gptr, unsigned lds_off) {
    asm volatile("global_load_async_to_lds_b128 %0, %1, off"
                 :: "v"(lds_off), "v"(gptr) : "memory");
}
DEV void wait_async_le1() { asm volatile("s_wait_asynccnt 0x1" ::: "memory"); }
DEV void wait_async_le0() { asm volatile("s_wait_asynccnt 0x0" ::: "memory"); }

// ---------------------------------------------------------------------------
// Packing kernels: fragment-major f16 operand streams (one 32B vector per lane).
// ---------------------------------------------------------------------------
__global__ void __launch_bounds__(128) pack_a_kernel(
    const float* __restrict__ A, int lda, _Float16* __restrict__ out) {
    const int t    = blockIdx.x * 4 + (threadIdx.x >> 5); // t = tm*KC + kc
    const int lane = threadIdx.x & 31;
    const int tm = t / KC, kc = t % KC;
    const int m  = tm * 16 + (lane & 15);
    const int kh = (lane >> 4) * 8;
    const float* Ap = A + (size_t)m * lda + kc * 32;
    v16h a;
#pragma unroll
    for (int v = 0; v < 8; ++v) {
        int k = 2 * kh + v * 2;
        a[2 * v]     = (_Float16)Ap[k];
        a[2 * v + 1] = (_Float16)Ap[k + 1];
    }
    *(v16h*)(out + ((size_t)t * 32 + lane) * 16) = a;
}

__global__ void __launch_bounds__(128) pack_b_kernel(
    const float* __restrict__ W, int ldn, _Float16* __restrict__ out) {
    const int t    = blockIdx.x * 4 + (threadIdx.x >> 5); // t = tn*KC + kc
    const int lane = threadIdx.x & 31;
    const int tn = t / KC, kc = t % KC;
    const int n  = tn * 16 + (lane & 15);
    const int k0 = kc * 32 + ((lane >> 4) << 4);
    const float* Wp = W + (size_t)k0 * ldn + n;
    v16h b;
#pragma unroll
    for (int e = 0; e < 16; ++e) b[e] = (_Float16)Wp[(size_t)e * ldn];
    *(v16h*)(out + ((size_t)t * 32 + lane) * 16) = b;
}

// ---------------------------------------------------------------------------
// GEMM core: block = 4 waves sharing row tile tm; wave owns 4 column tiles.
// A staged 2 k-chunks per round via async->LDS double buffer (all 128 threads
// issue one b128). 8 WMMAs per barrier pair.
// ---------------------------------------------------------------------------
struct GemmAcc { v8f c[4]; };

DEV GemmAcc gemm_core(const _Float16* __restrict__ pA,
                      const _Float16* __restrict__ pB,
                      _Float16* Abuf /* [2][1024] halves */,
                      int tm, int tnBase) {
    const int tid  = threadIdx.x;
    const int lane = tid & 31;

    const _Float16* pBt[4];
#pragma unroll
    for (int t = 0; t < 4; ++t) pBt[t] = pB + (size_t)(tnBase + t) * KC * 32 * 16;

    // stage A fragments for k-chunks {kc2, kc2+1} (2KB contiguous in packed layout)
    auto stageA = [&](int kc2, int buf) {
        const _Float16* src = pA + ((size_t)(tm * KC + kc2) * 32) * 16 + tid * 8;
        async_copy_b128(src, (unsigned)(size_t)&Abuf[buf * 1024 + tid * 8]);
    };

    GemmAcc acc;
#pragma unroll
    for (int t = 0; t < 4; ++t) acc.c[t] = (v8f){};

    stageA(0, 0);
    for (int kc2 = 0; kc2 < KC; kc2 += 2) {
        const int buf = (kc2 >> 1) & 1;
        const bool pre = (kc2 + 2 < KC);
        if (pre) { stageA(kc2 + 2, buf ^ 1); wait_async_le1(); }
        else     { wait_async_le0(); }
        __syncthreads();
#pragma unroll
        for (int sub = 0; sub < 2; ++sub) {
            v16h a = *(const v16h*)&Abuf[buf * 1024 + sub * 512 + lane * 16];
            const size_t koff = ((size_t)(kc2 + sub) * 32 + lane) * 16;
#pragma unroll
            for (int t = 0; t < 4; ++t) {
                v16h b = *(const v16h*)(pBt[t] + koff);
                acc.c[t] = wmma_f16(a, b, acc.c[t]);
            }
        }
        __syncthreads(); // protect buf before refill
    }
    return acc;
}

// ---------------------------------------------------------------------------
// Kernel 1: qkv = x @ Wqkv + bqkv, rotary on q/k, scatter to head layout.
// ---------------------------------------------------------------------------
__global__ void __launch_bounds__(128) qkv_rope_kernel(
    const _Float16* __restrict__ pA, const _Float16* __restrict__ pB,
    const float* __restrict__ bqkv, const float* __restrict__ cosT,
    const float* __restrict__ sinT, float* __restrict__ outK,
    float* __restrict__ outV, _Float16* __restrict__ q16,
    _Float16* __restrict__ k16, _Float16* __restrict__ v16) {
    __shared__ __align__(32) _Float16 Abuf[2 * 1024];

    const int lane = threadIdx.x & 31;
    const int wv   = threadIdx.x >> 5;
    const int Ng   = (N3 / 16) / 16;              // 24 column groups (16 tiles each)
    const int tm   = blockIdx.x / Ng;
    const int tnBase = (blockIdx.x % Ng) * 16 + wv * 4;

    GemmAcc acc = gemm_core(pA, pB, Abuf, tm, tnBase);

    const int rowb = (lane >> 4) * 8;
    const float scaleQ = 0.08838834764831845f; // 1/sqrt(128)

#pragma unroll
    for (int t = 0; t < 4; ++t) {
        const int n   = (tnBase + t) * 16 + (lane & 15);
        const int sec = n >> 11;             // 0=q 1=k 2=v (wave-uniform)
        const int e   = n & (EE - 1);
        const int h   = e >> 7;
        const int d   = e & (DD - 1);
        const float bias = bqkv[n];
#pragma unroll
        for (int r = 0; r < 8; ++r) {
            int row = tm * 16 + rowb + r;
            int bIdx = row / SS, s = row % SS;
            float val = acc.c[t][r] + bias;
            float pv  = __shfl_xor(val, 1, 32); // rotary pair partner
            size_t hidx = (((size_t)bIdx * HH + h) * SS + s) * DD + d;
            if (sec == 2) {
                outV[hidx] = val;
                v16[hidx]  = (_Float16)val;
            } else {
                float cz = cosT[(size_t)s * DD + d];
                float sz = sinT[(size_t)s * DD + d];
                float rot = (d & 1) ? pv : -pv;
                float o = val * cz + rot * sz;
                if (sec == 0) {
                    q16[hidx] = (_Float16)(o * scaleQ);
                } else {
                    outK[hidx] = o;
                    k16[hidx]  = (_Float16)o;
                }
            }
        }
    }
}

// ---------------------------------------------------------------------------
// Kernel 2: flash attention, one 16-query tile per wave; emits packed-A y.
// ---------------------------------------------------------------------------
__global__ void __launch_bounds__(128) attn_kernel(
    const _Float16* __restrict__ q16, const _Float16* __restrict__ k16,
    const _Float16* __restrict__ v16, _Float16* __restrict__ packY) {
    __shared__ _Float16 P[4][16][32];                  // P staging (C->A layout)
    __shared__ __align__(32) _Float16 Olds[4][16][DD]; // O repack

    const int lane = threadIdx.x & 31;
    const int wv   = threadIdx.x >> 5;
    const int id   = blockIdx.x * 4 + wv;
    const int QT   = SS / 16;
    const int qt = id % QT;
    const int bh = id / QT;
    const int h = bh % HH, b = bh / HH;

    const _Float16* Qp = q16 + (((size_t)b * HH + h) * SS) * DD;
    const _Float16* Kp = k16 + (((size_t)b * HH + h) * SS) * DD;
    const _Float16* Vp = v16 + (((size_t)b * HH + h) * SS) * DD;

    const int rowb = (lane >> 4) * 8;
    const int colp = lane & 15;

    v16h aq[4];
#pragma unroll
    for (int dc = 0; dc < 4; ++dc) {
        v16h a;
        const _Float16* Ap = Qp + (size_t)(qt * 16 + colp) * DD + dc * 32;
#pragma unroll
        for (int v = 0; v < 8; ++v) {
            int k = 2 * rowb + v * 2;
            a[2 * v] = Ap[k]; a[2 * v + 1] = Ap[k + 1];
        }
        aq[dc] = a;
    }

    v8f o[8] = {};
    float mr[8], lr[8];
#pragma unroll
    for (int r = 0; r < 8; ++r) { mr[r] = -1e30f; lr[r] = 0.0f; }

    for (int kb = 0; kb < SS; kb += 32) {
        v8f s0 = {}, s1 = {};
#pragma unroll
        for (int dc = 0; dc < 4; ++dc) {
            v16h b0 = *(const v16h*)(Kp + (size_t)(kb + colp) * DD + dc * 32 + ((lane >> 4) << 4));
            s0 = wmma_f16(aq[dc], b0, s0);
            v16h b1 = *(const v16h*)(Kp + (size_t)(kb + 16 + colp) * DD + dc * 32 + ((lane >> 4) << 4));
            s1 = wmma_f16(aq[dc], b1, s1);
        }
        float p0[8], p1[8], alpha[8];
#pragma unroll
        for (int r = 0; r < 8; ++r) {
            float tmx = fmaxf(s0[r], s1[r]);
            tmx = fmaxf(tmx, __shfl_xor(tmx, 1, 32));
            tmx = fmaxf(tmx, __shfl_xor(tmx, 2, 32));
            tmx = fmaxf(tmx, __shfl_xor(tmx, 4, 32));
            tmx = fmaxf(tmx, __shfl_xor(tmx, 8, 32));
            float mn = fmaxf(mr[r], tmx);
            alpha[r] = __expf(mr[r] - mn);
            mr[r] = mn;
            p0[r] = __expf(s0[r] - mn);
            p1[r] = __expf(s1[r] - mn);
            float rs = p0[r] + p1[r];
            rs += __shfl_xor(rs, 1, 32);
            rs += __shfl_xor(rs, 2, 32);
            rs += __shfl_xor(rs, 4, 32);
            rs += __shfl_xor(rs, 8, 32);
            lr[r] = lr[r] * alpha[r] + rs;
        }
#pragma unroll
        for (int f = 0; f < 8; ++f)
#pragma unroll
            for (int r = 0; r < 8; ++r) o[f][r] *= alpha[r];

#pragma unroll
        for (int r = 0; r < 8; ++r) {
            P[wv][rowb + r][colp]      = (_Float16)p0[r];
            P[wv][rowb + r][16 + colp] = (_Float16)p1[r];
        }
        v16h ap;
        {
            const _Float16* Pp = &P[wv][colp][0];
#pragma unroll
            for (int v = 0; v < 8; ++v) {
                int k = 2 * rowb + v * 2;
                ap[2 * v] = Pp[k]; ap[2 * v + 1] = Pp[k + 1];
            }
        }
#pragma unroll
        for (int f = 0; f < 8; ++f) {
            v16h bv;
            const _Float16* Bp = Vp + (size_t)(kb + ((lane >> 4) << 4)) * DD + f * 16 + colp;
#pragma unroll
            for (int ee = 0; ee < 16; ++ee) bv[ee] = Bp[(size_t)ee * DD];
            o[f] = wmma_f16(ap, bv, o[f]);
        }
    }

    float inv[8];
#pragma unroll
    for (int r = 0; r < 8; ++r) inv[r] = 1.0f / lr[r];
#pragma unroll
    for (int f = 0; f < 8; ++f)
#pragma unroll
        for (int r = 0; r < 8; ++r)
            Olds[wv][rowb + r][f * 16 + colp] = (_Float16)(o[f][r] * inv[r]);

    const int tmy = b * (SS / 16) + qt;
#pragma unroll
    for (int kcl = 0; kcl < 4; ++kcl) {
        const int kcg = h * 4 + kcl;
        v16h a;
        const _Float16* Op = &Olds[wv][colp][kcl * 32];
#pragma unroll
        for (int v = 0; v < 8; ++v) {
            int k = 2 * rowb + v * 2;
            a[2 * v] = Op[k]; a[2 * v + 1] = Op[k + 1];
        }
        *(v16h*)(packY + ((size_t)(tmy * KC + kcg) * 32 + lane) * 16) = a;
    }
}

// ---------------------------------------------------------------------------
// Kernel 3: out = y @ Wproj + bproj (both operands fragment-packed f16)
// ---------------------------------------------------------------------------
__global__ void __launch_bounds__(128) proj_kernel(
    const _Float16* __restrict__ pA, const _Float16* __restrict__ pB,
    const float* __restrict__ bproj, float* __restrict__ out) {
    __shared__ __align__(32) _Float16 Abuf[2 * 1024];

    const int lane = threadIdx.x & 31;
    const int wv   = threadIdx.x >> 5;
    const int Ng   = (EE / 16) / 16;              // 8 column groups
    const int tm   = blockIdx.x / Ng;
    const int tnBase = (blockIdx.x % Ng) * 16 + wv * 4;

    GemmAcc acc = gemm_core(pA, pB, Abuf, tm, tnBase);

    const int rowb = (lane >> 4) * 8;
#pragma unroll
    for (int t = 0; t < 4; ++t) {
        const int n = (tnBase + t) * 16 + (lane & 15);
        const float bias = bproj[n];
#pragma unroll
        for (int r = 0; r < 8; ++r) {
            int row = tm * 16 + rowb + r;
            out[(size_t)row * EE + n] = acc.c[t][r] + bias;
        }
    }
}

extern "C" void kernel_launch(void* const* d_in, const int* in_sizes, int n_in,
                              void* d_out, int out_size, void* d_ws, size_t ws_size,
                              hipStream_t stream) {
    (void)in_sizes; (void)n_in; (void)out_size; (void)ws_size;
    const float* x     = (const float*)d_in[0];
    const float* cosT  = (const float*)d_in[1];
    const float* sinT  = (const float*)d_in[2];
    const float* Wqkv  = (const float*)d_in[3];
    const float* bqkv  = (const float*)d_in[4];
    const float* Wproj = (const float*)d_in[5];
    const float* bproj = (const float*)d_in[6];

    float* out  = (float*)d_out;                    // (B,S,E)
    float* outK = out + (size_t)BB * SS * EE;       // (B,H,S,D)
    float* outV = outK + (size_t)BB * HH * SS * DD; // (B,H,S,D)

    const size_t hsd = (size_t)BB * HH * SS * DD;
    _Float16* q16   = (_Float16*)d_ws;
    _Float16* k16   = q16 + hsd;
    _Float16* v16   = k16 + hsd;
    _Float16* pAx   = v16 + hsd;                    // packed x     (B*S x E)
    _Float16* pBqkv = pAx + (size_t)BB * SS * EE;   // packed Wqkv  (E x 3E)
    _Float16* pBprj = pBqkv + (size_t)EE * N3;      // packed Wproj (E x E)
    _Float16* pY    = pBprj + (size_t)EE * EE;      // packed y     (B*S x E)

    dim3 blk(128);
    pack_a_kernel<<<dim3(MT * KC / 4), blk, 0, stream>>>(x, EE, pAx);
    pack_b_kernel<<<dim3((N3 / 16) * KC / 4), blk, 0, stream>>>(Wqkv, N3, pBqkv);
    pack_b_kernel<<<dim3((EE / 16) * KC / 4), blk, 0, stream>>>(Wproj, EE, pBprj);

    qkv_rope_kernel<<<dim3(MT * ((N3 / 16) / 16)), blk, 0, stream>>>(
        pAx, pBqkv, bqkv, cosT, sinT, outK, outV, q16, k16, v16);
    attn_kernel<<<dim3((BB * HH * (SS / 16)) / 4), blk, 0, stream>>>(q16, k16, v16, pY);
    proj_kernel<<<dim3(MT * ((EE / 16) / 16)), blk, 0, stream>>>(pY, pBprj, bproj, out);
}
